// JMF3_inv_22127671509715
// MI455X (gfx1250) — matrix operand order repaired and checked by reading, not compiled, not run on hardware
//
#include <hip/hip_runtime.h>
#include <hip/hip_bf16.h>

typedef _Float16 f16;
typedef __attribute__((ext_vector_type(8)))  _Float16 v8h;
typedef __attribute__((ext_vector_type(16))) _Float16 v16h;
typedef __attribute__((ext_vector_type(8)))  float    v8f;

#define NS   1024            // samples
#define NN   192             // sites
#define ND   24              // distance classes
#define NCOL (ND*NN)         // 4608 combined (d,j) columns
#define NCT  (NCOL/16)       // 288 column tiles of 16

// ---------------------------------------------------------------------------
// Kernel 1: build column-major f16 tables
//   E[col*192 + k] = W3[nd[k,j], d]     (col = d*192 + j)
//   F[col*192 + k] = (nd[k,j] == d)
// ---------------------------------------------------------------------------
__global__ __launch_bounds__(256) void build_tables(const float* __restrict__ W3,
                                                    const int*   __restrict__ nd,
                                                    f16* __restrict__ E,
                                                    f16* __restrict__ F) {
    int idx = blockIdx.x * blockDim.x + threadIdx.x;   // over NCOL*NN
    if (idx >= NN * NCOL) return;
    int col = idx / NN;
    int k   = idx - col * NN;
    int d   = col / NN;
    int j   = col - d * NN;
    int t   = nd[k * NN + j];
    E[idx] = (f16)W3[t * ND + d];
    F[idx] = (t == d) ? (f16)1.0f : (f16)0.0f;
}

// ---------------------------------------------------------------------------
// Kernel 2: x (f32, +-1) -> f16 row-major copy for A fragments
// ---------------------------------------------------------------------------
__global__ __launch_bounds__(256) void cvt_x(const float* __restrict__ x,
                                             f16* __restrict__ x16) {
    int idx = blockIdx.x * blockDim.x + threadIdx.x;
    if (idx < NS * NN) x16[idx] = (f16)x[idx];
}

// ---------------------------------------------------------------------------
// Fragment loaders (wave32, v_wmma_f32_16x16x32_f16 operand layouts)
// A (16x32, row-major source, row stride 192):
//   lane l: M = l%16, half = l/16
//   VGPR0..3 : K = K0 + half*8 + 0..7      -> one b128 load
//   VGPR4..7 : K = K0 + 16 + half*8 + 0..7 -> one b128 load
// B (32x16, column-major source = each column's 192 K-values contiguous):
//   lane l: N = l%16, half = l/16; K = K0 + half*8 + 0..7, then K0 + 16 + half*8 + 0..7
// ---------------------------------------------------------------------------
__device__ __forceinline__ v16h load_a_frag(const f16* __restrict__ row, int K0, int half) {
    const v8h lo = *(const v8h*)(row + K0 + half * 8);
    const v8h hi = *(const v8h*)(row + K0 + 16 + half * 8);
    return __builtin_shufflevector(lo, hi, 0,1,2,3,4,5,6,7,8,9,10,11,12,13,14,15);
}
__device__ __forceinline__ v16h load_b_frag(const f16* __restrict__ colp, int K0, int half) {
    const v8h lo = *(const v8h*)(colp + K0 + half * 8);
    const v8h hi = *(const v8h*)(colp + K0 + 16 + half * 8);
    return __builtin_shufflevector(lo, hi, 0,1,2,3,4,5,6,7,8,9,10,11,12,13,14,15);
}

// ---------------------------------------------------------------------------
// Kernel 3: main WMMA kernel.
// One wave = one 16(sample) x 16(col) tile: S-tile and c-tile via 6+6 WMMAs,
// fused epilogue  partial = sum_cols  x[s,j] * c * (S + 0.5*W2[d]),
// cross-lane reduced and written to partial[ct][s].
// grid = (NS/16, NCT/8), block = 256 (8 waves, wave w -> col tile blockIdx.y*8+w)
// ---------------------------------------------------------------------------
__global__ __launch_bounds__(256) void jmf3_wmma(const f16*   __restrict__ x16,
                                                 const float* __restrict__ x,
                                                 const f16*   __restrict__ E,
                                                 const f16*   __restrict__ F,
                                                 const float* __restrict__ W2,
                                                 float*       __restrict__ partial) {
    const int lane = threadIdx.x & 31;
    const int wave = threadIdx.x >> 5;
    const int r    = lane & 15;       // column-within-tile (B/N index) / row (A/M index)
    const int half = lane >> 4;

    const int s0 = blockIdx.x * 16;               // sample tile base
    const int ct = blockIdx.y * 8 + wave;         // column tile 0..287
    const int d  = ct / (NN / 16);                // 192/16=12 tiles per d-band
    const int j0 = (ct % (NN / 16)) * 16;         // site base for this tile
    const int c  = ct * 16 + r;                   // this lane's column

    const f16* arow  = x16 + (size_t)(s0 + r) * NN;
    const f16* bcolE = E + (size_t)c * NN;
    const f16* bcolF = F + (size_t)c * NN;

    // Preload full A fragments for K = 0..191 (6 chunks of 32)
    v16h afrag[6];
#pragma unroll
    for (int kc = 0; kc < 6; ++kc) afrag[kc] = load_a_frag(arow, kc * 32, half);

    v8f accS = {};   // S tile  (x @ E)
    v8f accC = {};   // c tile  (x @ F)
#pragma unroll
    for (int kc = 0; kc < 6; ++kc) {
        v16h bE = load_b_frag(bcolE, kc * 32, half);
        accS = __builtin_amdgcn_wmma_f32_16x16x32_f16(false, afrag[kc], false, bE,
                                                      (short)0, accS, false, false);
        v16h bF = load_b_frag(bcolF, kc * 32, half);
        accC = __builtin_amdgcn_wmma_f32_16x16x32_f16(false, afrag[kc], false, bF,
                                                      (short)0, accC, false, false);
    }

    const float hw2 = 0.5f * W2[d];

    // Epilogue: element v of C/D tile is (M = v + 8*half, N = r)
    float red[8];
#pragma unroll
    for (int v = 0; v < 8; ++v) {
        const int m = v + half * 8;
        const float xj = x[(size_t)(s0 + m) * NN + j0 + r];
        red[v] = accC[v] * (accS[v] + hw2) * xj;
    }
    // Sum across the 16 columns (lanes with same `half`: xor masks 1,2,4,8)
#pragma unroll
    for (int mask = 1; mask < 16; mask <<= 1) {
#pragma unroll
        for (int v = 0; v < 8; ++v) red[v] += __shfl_xor(red[v], mask, 32);
    }
    if (r == 0) {
#pragma unroll
        for (int v = 0; v < 8; ++v)
            partial[(size_t)ct * NS + s0 + v + half * 8] = red[v];
    }
}

// ---------------------------------------------------------------------------
// Kernel 4: out[s] = sum_i log(phi[i, x>0]) + sum_ct partial[ct][s]
// Fixed summation order -> deterministic output.
// ---------------------------------------------------------------------------
__global__ __launch_bounds__(256) void finalize(const float* __restrict__ x,
                                                const float* __restrict__ phi,
                                                const float* __restrict__ partial,
                                                float* __restrict__ out) {
    int s = blockIdx.x * blockDim.x + threadIdx.x;
    if (s >= NS) return;
    float lm = 0.0f;
#pragma unroll 4
    for (int i = 0; i < NN; ++i) {
        int b = (x[(size_t)s * NN + i] > 0.0f) ? 1 : 0;
        lm += logf(phi[i * 2 + b]);
    }
    float acc = 0.0f;
    for (int ct = 0; ct < NCT; ++ct) acc += partial[(size_t)ct * NS + s];
    out[s] = lm + acc;
}

// ---------------------------------------------------------------------------
// Launch
// ---------------------------------------------------------------------------
extern "C" void kernel_launch(void* const* d_in, const int* in_sizes, int n_in,
                              void* d_out, int out_size, void* d_ws, size_t ws_size,
                              hipStream_t stream) {
    const float* x   = (const float*)d_in[0];   // (1024,192)
    const float* W2  = (const float*)d_in[1];   // (24,)
    const float* W3  = (const float*)d_in[2];   // (24,24)
    const float* phi = (const float*)d_in[3];   // (192,2)
    const int*   nd  = (const int*)  d_in[4];   // (192,192)
    float* out = (float*)d_out;                 // (1024,)

    // Workspace layout (16B aligned slices)
    char* ws = (char*)d_ws;
    const size_t szEF = (size_t)NN * NCOL * sizeof(f16);   // 1,769,472 B each
    f16*   E       = (f16*)(ws);
    f16*   F       = (f16*)(ws + szEF);
    f16*   x16     = (f16*)(ws + 2 * szEF);
    float* partial = (float*)(ws + 2 * szEF + (size_t)NS * NN * sizeof(f16));

    build_tables<<<(NN * NCOL + 255) / 256, 256, 0, stream>>>(W3, nd, E, F);
    cvt_x<<<(NS * NN + 255) / 256, 256, 0, stream>>>(x, x16);
    jmf3_wmma<<<dim3(NS / 16, NCT / 8), 256, 0, stream>>>(x16, x, E, F, W2, partial);
    finalize<<<(NS + 255) / 256, 256, 0, stream>>>(x, phi, partial, out);
}
